// EGCL_2070174237291
// MI455X (gfx1250) — matrix operand order repaired and checked by reading, not compiled, not run on hardware
//
#include <hip/hip_runtime.h>

typedef __attribute__((ext_vector_type(16))) _Float16 v16h;
typedef __attribute__((ext_vector_type(8)))  _Float16 v8h;
typedef __attribute__((ext_vector_type(8)))  float    v8f;
typedef __attribute__((ext_vector_type(4)))  int      v4i;

#define DFEAT   128
#define KPAD_E  288   // 257 padded up to 9*32
#define KPAD_N  256   // 2*D, multiple of 32
#define ROWS    64    // edges / nodes per workgroup (4 row-tiles per wave)

// LDS row strides in halves, padded off the 64-bank pattern
#define AS_E    296   // edge A panel (288 + 8)
#define AS_N    264   // node A panel (256 + 8)
#define MS      136   // layer-1 activation panel (128 + 8)

__device__ __forceinline__ v16h cat8(v8h lo, v8h hi) {
  return __builtin_shufflevector(lo, hi, 0,1,2,3,4,5,6,7,8,9,10,11,12,13,14,15);
}

__device__ __forceinline__ float silu_f(float x) {
  return x * (1.0f / (1.0f + __expf(-x)));
}

// 16-byte global->LDS copy: async DMA path when available (ASYNCcnt-tracked),
// VGPR bounce otherwise.
__device__ __forceinline__ void copy16_g2l(const _Float16* g, _Float16* l) {
#if __has_builtin(__builtin_amdgcn_global_load_async_to_lds_b128)
  __builtin_amdgcn_global_load_async_to_lds_b128(
      (__attribute__((address_space(1))) v4i*)(v4i*)(void*)(_Float16*)g,
      (__attribute__((address_space(3))) v4i*)(v4i*)(void*)l,
      0, 0);
#else
  *(v8h*)l = *(const v8h*)g;
#endif
}

__device__ __forceinline__ void async_wait0() {
#if __has_builtin(__builtin_amdgcn_global_load_async_to_lds_b128)
#if __has_builtin(__builtin_amdgcn_s_wait_asynccnt)
  __builtin_amdgcn_s_wait_asynccnt(0);
#else
  asm volatile("s_wait_asynccnt 0" ::: "memory");
#endif
#endif
}

// ---------------------------------------------------------------------------
// prep kernels
// ---------------------------------------------------------------------------
__global__ void zero_f32(float* __restrict__ p, int n) {
  int i = blockIdx.x * blockDim.x + threadIdx.x;
  int stride = gridDim.x * blockDim.x;
  for (; i < n; i += stride) p[i] = 0.0f;
}

__global__ void f32_to_f16(const float* __restrict__ s, _Float16* __restrict__ d, int n) {
  int i = blockIdx.x * blockDim.x + threadIdx.x;
  int stride = gridDim.x * blockDim.x;
  for (; i < n; i += stride) d[i] = (_Float16)s[i];
}

// WT[n][k] = (k < Ksrc) ? W[k][n] : 0   (f32 [Ksrc x 128] -> f16 [128 x Kpad])
__global__ void convT_f16(const float* __restrict__ W, _Float16* __restrict__ WT,
                          int Ksrc, int Kpad) {
  int i = blockIdx.x * blockDim.x + threadIdx.x;
  int total = DFEAT * Kpad;
  if (i >= total) return;
  int n = i / Kpad;
  int k = i - n * Kpad;
  WT[(size_t)n * Kpad + k] = (k < Ksrc) ? (_Float16)W[(size_t)k * DFEAT + n]
                                        : (_Float16)0.0f;
}

// ---------------------------------------------------------------------------
// edge pass: 64 edges per 256-thread block; each wave owns a 16-col tile and
// sweeps 4 row-tiles, reusing each weight fragment 4x.
// ---------------------------------------------------------------------------
__global__ __launch_bounds__(256)
void egcl_edge(const _Float16* __restrict__ h16,
               const float* __restrict__ coord,
               const float* __restrict__ emask,
               const long long* __restrict__ eidx,
               const float* __restrict__ be1,
               const float* __restrict__ be2,
               const _Float16* __restrict__ Wt1,
               const _Float16* __restrict__ Wt2,
               float* __restrict__ agg,
               int E)
{
  __shared__ _Float16 Ain[ROWS * AS_E];   // gathered edge inputs, f16
  __shared__ _Float16 M1 [ROWS * MS];     // layer-1 activations, f16
  __shared__ float    smask[ROWS];
  __shared__ int      srow [ROWS];

  const int t  = threadIdx.x;
  const int eb = blockIdx.x * ROWS;

  // ---- radial / mask / row index / K padding (DS writes) ----
  if (t < ROWS) {
    int e  = eb + t;
    int ec = (e < E) ? e : (E - 1);
    long long r = eidx[ec];
    long long c = eidx[(long long)E + ec];
    srow[t] = (int)r;
    float dx = coord[r * 3 + 0] - coord[c * 3 + 0];
    float dy = coord[r * 3 + 1] - coord[c * 3 + 1];
    float dz = coord[r * 3 + 2] - coord[c * 3 + 2];
    float rad = dx * dx + dy * dy + dz * dz;
    Ain[t * AS_E + 256] = (_Float16)rad;
    #pragma unroll
    for (int k = 257; k < KPAD_E; ++k) Ain[t * AS_E + k] = (_Float16)0.0f;
    smask[t] = (e < E) ? emask[e] : 0.0f;
  }

  // ---- gather h16[row] || h16[col] via 16B global->LDS copies ----
  // 64 edges x 32 chunks (16 for row, 16 for col) x 16B = 2048 copies, 8/thread
  #pragma unroll
  for (int i = 0; i < 8; ++i) {
    int g   = i * 256 + t;
    int el  = g >> 5;          // edge within tile
    int sub = g & 31;          // 0..15 -> row source, 16..31 -> col source
    int e   = eb + el;
    int ec  = (e < E) ? e : (E - 1);
    long long src = (sub < 16) ? eidx[ec] : eidx[(long long)E + ec];
    int c = sub & 15;          // 16B chunk (8 halves)
    copy16_g2l(h16 + src * DFEAT + c * 8,
               &Ain[el * AS_E + ((sub < 16) ? 0 : DFEAT) + c * 8]);
  }
  async_wait0();
  __syncthreads();

  const int lane = t & 31;
  const int w    = t >> 5;
  const int hs   = lane >> 4;              // which K-half this lane holds
  const int ln   = lane & 15;              // row (A) / column (B,C) index
  const int gcol = w * 16 + ln;            // global output feature

  // ---- layer 1: [64 x 288] x [288 x 128], f16 WMMA, f32 accum ----
  v8f acc[4] = {{}, {}, {}, {}};
  const _Float16* Bcol = Wt1 + (size_t)gcol * KPAD_E + hs * 16;
  __builtin_prefetch(Bcol, 0, 1);
  #pragma unroll
  for (int kc = 0; kc < KPAD_E / 32; ++kc) {
    int k0 = kc * 32;
    v16h b = cat8(*(const v8h*)(Bcol + k0), *(const v8h*)(Bcol + k0 + 8));
    #pragma unroll
    for (int rt = 0; rt < 4; ++rt) {
      const _Float16* Arow = Ain + (rt * 16 + ln) * AS_E + hs * 16;
      v16h a = cat8(*(const v8h*)(Arow + k0), *(const v8h*)(Arow + k0 + 8));
      acc[rt] = __builtin_amdgcn_wmma_f32_16x16x32_f16(
          false, a, false, b, (short)0, acc[rt], false, false);
    }
  }
  {
    float b1 = be1[gcol];
    #pragma unroll
    for (int rt = 0; rt < 4; ++rt)
      #pragma unroll
      for (int j = 0; j < 8; ++j) {
        int r = rt * 16 + j + hs * 8;      // C layout: row j (+8 for hi lanes)
        M1[r * MS + gcol] = (_Float16)silu_f(acc[rt][j] + b1);
      }
  }
  __syncthreads();

  // ---- layer 2: [64 x 128] x [128 x 128] ----
  v8f acc2[4] = {{}, {}, {}, {}};
  const _Float16* Bcol2 = Wt2 + (size_t)gcol * DFEAT + hs * 16;
  #pragma unroll
  for (int kc = 0; kc < DFEAT / 32; ++kc) {
    int k0 = kc * 32;
    v16h b = cat8(*(const v8h*)(Bcol2 + k0), *(const v8h*)(Bcol2 + k0 + 8));
    #pragma unroll
    for (int rt = 0; rt < 4; ++rt) {
      const _Float16* Arow = M1 + (rt * 16 + ln) * MS + hs * 16;
      v16h a = cat8(*(const v8h*)(Arow + k0), *(const v8h*)(Arow + k0 + 8));
      acc2[rt] = __builtin_amdgcn_wmma_f32_16x16x32_f16(
          false, a, false, b, (short)0, acc2[rt], false, false);
    }
  }
  {
    float b2 = be2[gcol];
    #pragma unroll
    for (int rt = 0; rt < 4; ++rt)
      #pragma unroll
      for (int j = 0; j < 8; ++j) {
        int r = rt * 16 + j + hs * 8;
        int e = eb + r;
        float val = silu_f(acc2[rt][j] + b2) * smask[r];
        if (e < E) {
          __hip_atomic_fetch_add(&agg[(size_t)srow[r] * DFEAT + gcol], val,
                                 __ATOMIC_RELAXED, __HIP_MEMORY_SCOPE_AGENT);
        }
      }
  }
}

// ---------------------------------------------------------------------------
// node pass: 64 nodes per 256-thread block
// ---------------------------------------------------------------------------
__global__ __launch_bounds__(256)
void egcl_node(const float* __restrict__ h,
               const _Float16* __restrict__ h16,
               const float* __restrict__ agg,
               const float* __restrict__ bn1,
               const float* __restrict__ bn2,
               const _Float16* __restrict__ Wt1,
               const _Float16* __restrict__ Wt2,
               float* __restrict__ out,
               int N)
{
  __shared__ _Float16 Ain[ROWS * AS_N];
  __shared__ _Float16 M1 [ROWS * MS];

  const int t  = threadIdx.x;
  const int nb = blockIdx.x * ROWS;

  // ---- h half: 16B async copies from pre-converted h16 ----
  // 64 nodes x 16 chunks x 16B = 1024 copies, 4 per thread
  #pragma unroll
  for (int i = 0; i < 4; ++i) {
    int g  = i * 256 + t;
    int nl = g >> 4;
    int c  = g & 15;
    int node = nb + nl;
    int nc   = (node < N) ? node : (N - 1);
    copy16_g2l(h16 + (size_t)nc * DFEAT + c * 8, &Ain[nl * AS_N + c * 8]);
  }
  // ---- agg half: f32 -> f16 conversion through VALU ----
  // 64 nodes x 8 groups of 16 features = 512 groups, 2 per thread
  #pragma unroll
  for (int i = 0; i < 2; ++i) {
    int g  = i * 256 + t;
    int nl = g >> 3;
    int c  = g & 7;
    int node = nb + nl;
    int nc   = (node < N) ? node : (N - 1);
    const float4* p = (const float4*)(agg + (size_t)nc * DFEAT + c * 16);
    _Float16* dst = &Ain[nl * AS_N + DFEAT + c * 16];
    #pragma unroll
    for (int q = 0; q < 4; ++q) {
      float4 v = p[q];
      dst[q * 4 + 0] = (_Float16)v.x;
      dst[q * 4 + 1] = (_Float16)v.y;
      dst[q * 4 + 2] = (_Float16)v.z;
      dst[q * 4 + 3] = (_Float16)v.w;
    }
  }
  async_wait0();
  __syncthreads();

  const int lane = t & 31;
  const int w    = t >> 5;
  const int hs   = lane >> 4;
  const int ln   = lane & 15;
  const int gcol = w * 16 + ln;

  // ---- layer 1: [64 x 256] x [256 x 128] ----
  v8f acc[4] = {{}, {}, {}, {}};
  const _Float16* Bcol = Wt1 + (size_t)gcol * KPAD_N + hs * 16;
  __builtin_prefetch(Bcol, 0, 1);
  #pragma unroll
  for (int kc = 0; kc < KPAD_N / 32; ++kc) {
    int k0 = kc * 32;
    v16h b = cat8(*(const v8h*)(Bcol + k0), *(const v8h*)(Bcol + k0 + 8));
    #pragma unroll
    for (int rt = 0; rt < 4; ++rt) {
      const _Float16* Arow = Ain + (rt * 16 + ln) * AS_N + hs * 16;
      v16h a = cat8(*(const v8h*)(Arow + k0), *(const v8h*)(Arow + k0 + 8));
      acc[rt] = __builtin_amdgcn_wmma_f32_16x16x32_f16(
          false, a, false, b, (short)0, acc[rt], false, false);
    }
  }
  {
    float b1 = bn1[gcol];
    #pragma unroll
    for (int rt = 0; rt < 4; ++rt)
      #pragma unroll
      for (int j = 0; j < 8; ++j) {
        int r = rt * 16 + j + hs * 8;
        M1[r * MS + gcol] = (_Float16)silu_f(acc[rt][j] + b1);
      }
  }
  __syncthreads();

  // ---- layer 2 + residual ----
  v8f acc2[4] = {{}, {}, {}, {}};
  const _Float16* Bcol2 = Wt2 + (size_t)gcol * DFEAT + hs * 16;
  #pragma unroll
  for (int kc = 0; kc < DFEAT / 32; ++kc) {
    int k0 = kc * 32;
    v16h b = cat8(*(const v8h*)(Bcol2 + k0), *(const v8h*)(Bcol2 + k0 + 8));
    #pragma unroll
    for (int rt = 0; rt < 4; ++rt) {
      const _Float16* Arow = M1 + (rt * 16 + ln) * MS + hs * 16;
      v16h a = cat8(*(const v8h*)(Arow + k0), *(const v8h*)(Arow + k0 + 8));
      acc2[rt] = __builtin_amdgcn_wmma_f32_16x16x32_f16(
          false, a, false, b, (short)0, acc2[rt], false, false);
    }
  }
  {
    float b2 = bn2[gcol];
    #pragma unroll
    for (int rt = 0; rt < 4; ++rt)
      #pragma unroll
      for (int j = 0; j < 8; ++j) {
        int r = rt * 16 + j + hs * 8;
        int node = nb + r;
        if (node < N) {
          size_t o = (size_t)node * DFEAT + gcol;
          out[o] = h[o] + acc2[rt][j] + b2;
        }
      }
  }
}

// ---------------------------------------------------------------------------
// host launcher
// ---------------------------------------------------------------------------
extern "C" void kernel_launch(void* const* d_in, const int* in_sizes, int n_in,
                              void* d_out, int out_size, void* d_ws, size_t ws_size,
                              hipStream_t stream)
{
  const float*     h     = (const float*)d_in[0];
  const float*     coord = (const float*)d_in[1];
  const float*     emask = (const float*)d_in[2];
  const float*     We1   = (const float*)d_in[3];
  const float*     be1   = (const float*)d_in[4];
  const float*     We2   = (const float*)d_in[5];
  const float*     be2   = (const float*)d_in[6];
  const float*     Wn1   = (const float*)d_in[7];
  const float*     bn1   = (const float*)d_in[8];
  const float*     Wn2   = (const float*)d_in[9];
  const float*     bn2   = (const float*)d_in[10];
  const long long* eidx  = (const long long*)d_in[11];

  const int N = in_sizes[0] / DFEAT;
  const int E = in_sizes[2];

  // workspace carve-out (256B aligned)
  char* ws = (char*)d_ws;
  size_t off = 0;
  auto carve = [&](size_t bytes) -> void* {
    void* p = ws + off;
    off = (off + bytes + 255) & ~(size_t)255;
    return p;
  };
  float*    agg  = (float*)   carve((size_t)N * DFEAT * sizeof(float));
  _Float16* h16  = (_Float16*)carve((size_t)N * DFEAT * sizeof(_Float16));
  _Float16* Wt1e = (_Float16*)carve((size_t)DFEAT * KPAD_E * sizeof(_Float16));
  _Float16* Wt2e = (_Float16*)carve((size_t)DFEAT * DFEAT  * sizeof(_Float16));
  _Float16* Wt1n = (_Float16*)carve((size_t)DFEAT * KPAD_N * sizeof(_Float16));
  _Float16* Wt2n = (_Float16*)carve((size_t)DFEAT * DFEAT  * sizeof(_Float16));

  // 1) zero the scatter accumulator; pre-convert h to f16
  {
    int n = N * DFEAT;
    int blocks = (n + 255) / 256;
    if (blocks > 1024) blocks = 1024;
    zero_f32 <<<blocks, 256, 0, stream>>>(agg, n);
    f32_to_f16<<<blocks, 256, 0, stream>>>(h, h16, n);
  }
  // 2) transpose-convert weights to f16 [out_col][K] with K padding
  convT_f16<<<(DFEAT * KPAD_E + 255) / 256, 256, 0, stream>>>(We1, Wt1e, 257, KPAD_E);
  convT_f16<<<(DFEAT * DFEAT  + 255) / 256, 256, 0, stream>>>(We2, Wt2e, DFEAT, DFEAT);
  convT_f16<<<(DFEAT * KPAD_N + 255) / 256, 256, 0, stream>>>(Wn1, Wt1n, KPAD_N, KPAD_N);
  convT_f16<<<(DFEAT * DFEAT  + 255) / 256, 256, 0, stream>>>(Wn2, Wt2n, DFEAT, DFEAT);

  // 3) edge MLP + scatter-add (~31.5 GFLOP, WMMA)
  {
    int blocks = (E + ROWS - 1) / ROWS;
    egcl_edge<<<blocks, 256, 0, stream>>>(h16, coord, emask, eidx, be1, be2,
                                          Wt1e, Wt2e, agg, E);
  }
  // 4) node MLP + residual
  {
    int blocks = (N + ROWS - 1) / ROWS;
    egcl_node<<<blocks, 256, 0, stream>>>(h, h16, agg, bn1, bn2, Wt1n, Wt2n,
                                          (float*)d_out, N);
  }
}
